// VanillaRNN_66443144069543
// MI455X (gfx1250) — compile-verified, matrix operand order checked by reading, not run on hardware
//
#include <hip/hip_runtime.h>
#include <hip/hip_fp16.h>

typedef __attribute__((ext_vector_type(16))) _Float16 v16h;
typedef __attribute__((ext_vector_type(8)))  float    v8f;

#define SEQ 512
#define NB  4096
#define NH  2048
#define NO  10
#define JPL 64   /* hidden units per lane = NH / 32 (wave32) */

#ifndef __has_builtin
#define __has_builtin(x) 0
#endif

static __device__ __forceinline__ float fast_tanh(float x) {
#if __has_builtin(__builtin_amdgcn_tanhf)
  return __builtin_amdgcn_tanhf(x);
#else
  float r;
  asm("v_tanh_f32 %0, %1" : "=v"(r) : "v"(x));
  return r;
#endif
}

// Load 8 consecutive floats (32B-aligned) into a register array slice.
#define LD8(dst, base)                                        \
  {                                                           \
    const float4 t0 = *(const float4*)(base);                 \
    const float4 t1 = *(const float4*)((base) + 4);           \
    (dst)[0] = t0.x; (dst)[1] = t0.y; (dst)[2] = t0.z;        \
    (dst)[3] = t0.w; (dst)[4] = t1.x; (dst)[5] = t1.y;        \
    (dst)[6] = t1.z; (dst)[7] = t1.w;                         \
  }

// ---------------------------------------------------------------------------
// Kernel 1: 512-step scalar recurrence (rank-1 RNN collapse).
// One wave per batch element; each lane owns 64 contiguous hidden units held
// in VGPRs. Per step: 64 x (v_tanh_f32 + FMAs) per lane, wave32 butterfly
// reduction, scalar state update. Trans-pipe bound by design.
// ---------------------------------------------------------------------------
__global__ __launch_bounds__(256) void rnn_scan_kernel(
    const float* __restrict__ x, const float* __restrict__ Wx,
    const float* __restrict__ bx, const float* __restrict__ Wh,
    const float* __restrict__ bh, float* __restrict__ g)
{
  const int lane = threadIdx.x & 31;
  const int b = blockIdx.x * 8 + (threadIdx.x >> 5);

  // Each lane holds a contiguous block of 64 hidden units -> b128 preload.
  float wxr[JPL], bxr[JPL], whr[JPL];
  const int jb = lane * JPL;
#pragma unroll
  for (int k = 0; k < JPL; k += 8) {
    LD8(&wxr[k], Wx + jb + k);
    LD8(&bxr[k], bx + jb + k);
    LD8(&whr[k], Wh + jb + k);
  }

  const float bh0 = bh[0];
  float s = bh0;                          // g_0 = bh
  const float* xp = x + b;
  float u = xp[0];

#pragma unroll 1
  for (int t = 0; t < SEQ - 1; ++t) {
    const float un = xp[(t + 1) * NB];    // prefetch next timestep scalar
    float a = 0.f;
#pragma unroll
    for (int k = 0; k < JPL; ++k) {
      const float arg = __builtin_fmaf(u, wxr[k], bxr[k]) + s;
      a = __builtin_fmaf(whr[k], fast_tanh(arg), a);
    }
#pragma unroll
    for (int off = 16; off > 0; off >>= 1)  // wave32 butterfly reduction
      a += __shfl_xor(a, off, 32);
    s = a + bh0;                          // g_{t+1}
    u = un;
  }
  if (lane == 0) g[b] = s;                // g_511
}

// ---------------------------------------------------------------------------
// Kernel 2: classifier head + softmax via WMMA. One wave per 16 batch rows.
// A = h_final tile (16x32 f16, regenerated from u_511/g_511 on the fly),
// B = Wy^T tile (32x16 f16, branch-free zero-pad from 10 to 16 cols),
// 64 chunks of v_wmma_f32_16x16x32_f16 accumulate the logits.
// ---------------------------------------------------------------------------
__global__ __launch_bounds__(256) void rnn_head_kernel(
    const float* __restrict__ x, const float* __restrict__ Wx,
    const float* __restrict__ bx, const float* __restrict__ g,
    const float* __restrict__ Wy, const float* __restrict__ by,
    float* __restrict__ out)
{
  const int lane = threadIdx.x & 31;
  const int wave = threadIdx.x >> 5;
  const int tile = blockIdx.x * 8 + wave;   // 256 tiles of 16 batch rows
  const int b0 = tile * 16;
  const int half = lane >> 4;
  const int q = lane & 15;                  // M row for A; N col for B/C/D

  const float uM = x[(SEQ - 1) * NB + b0 + q];
  const float gM = g[b0 + q];

  // Branch-free zero padding of the 10 output columns to 16.
  const int   nc  = (q < NO) ? q : 0;       // clamped row of Wy
  const float msk = (q < NO) ? 1.f : 0.f;
  const float byN = by[nc] * msk;
  const float* __restrict__ wyr = Wy + nc * NH;

  v8f acc = {};
#pragma unroll 1
  for (int kc = 0; kc < NH / 32; ++kc) {
    // ISA 16-bit 16x32 fragment layout: this lane's 16 K-values are two
    // contiguous 8-float runs at kb + 8*half and kb + 16 + 8*half.
    const int j0 = kc * 32 + half * 8;
    const int j1 = j0 + 16;

    float wxl[16], bxl[16], wyl[16];
    LD8(&wxl[0], Wx + j0);  LD8(&wxl[8], Wx + j1);
    LD8(&bxl[0], bx + j0);  LD8(&bxl[8], bx + j1);
    LD8(&wyl[0], wyr + j0); LD8(&wyl[8], wyr + j1);

    v16h af, bf;
#pragma unroll
    for (int e = 0; e < 16; ++e) {
      const float hv = fast_tanh(__builtin_fmaf(uM, wxl[e], bxl[e]) + gM);
      af[e] = (_Float16)hv;
      bf[e] = (_Float16)(wyl[e] * msk);
    }
    acc = __builtin_amdgcn_wmma_f32_16x16x32_f16(
        /*neg_a=*/false, af, /*neg_b=*/false, bf,
        /*c_mod=*/(short)0, acc, /*reuse_a=*/false, /*reuse_b=*/false);
  }

  // acc[r] = logits[M = r + 8*half][N = q]; softmax across N.
  // XOR masks 8,4,2,1 never cross a 16-lane group, so full-width (32)
  // shuffles give group-local reductions without sub-width clamp codegen.
#pragma unroll
  for (int r = 0; r < 8; ++r) {
    const float v = acc[r] + byN;
    float vm = (q < NO) ? v : -3.0e38f;
#pragma unroll
    for (int off = 8; off > 0; off >>= 1) {
      const float o = __shfl_xor(vm, off, 32);
      vm = vm > o ? vm : o;
    }
    float ev = (q < NO) ? __expf(v - vm) : 0.f;
    float sum = ev;
#pragma unroll
    for (int off = 8; off > 0; off >>= 1)
      sum += __shfl_xor(sum, off, 32);
    if (q < NO)
      out[(b0 + r + half * 8) * NO + q] = ev / sum;
  }
}

extern "C" void kernel_launch(void* const* d_in, const int* in_sizes, int n_in,
                              void* d_out, int out_size, void* d_ws, size_t ws_size,
                              hipStream_t stream) {
  (void)in_sizes; (void)n_in; (void)out_size; (void)ws_size;
  const float* x  = (const float*)d_in[0];
  const float* Wx = (const float*)d_in[1];
  const float* bx = (const float*)d_in[2];
  const float* Wh = (const float*)d_in[3];
  const float* bh = (const float*)d_in[4];
  const float* Wy = (const float*)d_in[5];
  const float* by = (const float*)d_in[6];
  float* g   = (float*)d_ws;      // 4096 floats of scratch: g_511 per batch
  float* out = (float*)d_out;     // (4096, 10) f32

  rnn_scan_kernel<<<NB / 8, 256, 0, stream>>>(x, Wx, bx, Wh, bh, g);
  rnn_head_kernel<<<NB / 16 / 8, 256, 0, stream>>>(x, Wx, bx, g, Wy, by, out);
}
